// GCNNet_53970559042042
// MI455X (gfx1250) — matrix operand order repaired and checked by reading, not compiled, not run on hardware
//
#include <hip/hip_runtime.h>

typedef __attribute__((ext_vector_type(16))) __bf16 v16bf;
typedef __attribute__((ext_vector_type(8)))  float  v8f;

union Frag { v16bf v; uint4 u[2]; unsigned short s[16]; };

__device__ __forceinline__ unsigned short f2bf(float f) {
  unsigned int u = __float_as_uint(f);
  u += 0x7FFFu + ((u >> 16) & 1u);   // round-to-nearest-even
  return (unsigned short)(u >> 16);
}

// ---------------------------------------------------------------------------
// WMMA GEMM on pre-swizzled fragment-ready bf16 operands.
// Asw: [Mtiles][KC][32 lanes][16 bf16]  (A-matrix 16x32 fragment layout)
// Bsw: [Ntiles][KC][32 lanes][16 bf16]  (B-matrix 32x16 fragment layout)
// Each wave computes one 16x16 f32 tile of C; 4 waves per block.
// Inner loop: 4x global_load_b128 + 1x v_wmma_f32_16x16x32_bf16. No LDS.
// ---------------------------------------------------------------------------
__global__ __launch_bounds__(128)
void wmma_gemm_kernel(const unsigned short* __restrict__ Asw,
                      const unsigned short* __restrict__ Bsw,
                      const float* __restrict__ bias,
                      float* __restrict__ C, int ldc,
                      int Mtiles, int Ntiles, int KC,
                      int Nn, int relu_out)
{
  const int lane = threadIdx.x;
  const int mt = blockIdx.x * 4 + threadIdx.y;
  const int nt = blockIdx.y;
  if (mt >= Mtiles) return;          // uniform per-wave branch; EXEC stays ~0xffffffff

  const unsigned short* Ap = Asw + (long)mt * KC * 512 + lane * 16;
  const unsigned short* Bp = Bsw + (long)nt * KC * 512 + lane * 16;

  v8f acc = {0.f,0.f,0.f,0.f,0.f,0.f,0.f,0.f};
  for (int kc = 0; kc < KC; ++kc) {
    Frag a, b;
    a.u[0] = *(const uint4*)(Ap);
    a.u[1] = *(const uint4*)(Ap + 8);
    b.u[0] = *(const uint4*)(Bp);
    b.u[1] = *(const uint4*)(Bp + 8);
    acc = __builtin_amdgcn_wmma_f32_16x16x32_bf16(false, a.v, false, b.v,
                                                  (short)0, acc, false, false);
    Ap += 512;
    Bp += 512;
  }

  // C/D layout: VGPR r -> M = r + 8*(lane>>4), N = lane&15 (Mtiles exact, no m guard)
  const int half = lane >> 4, q = lane & 15;
  const int n = nt * 16 + q;
  if (n < Nn) {
    const float bv = bias ? bias[n] : 0.0f;
#pragma unroll
    for (int r = 0; r < 8; ++r) {
      const int m = mt * 16 + r + 8 * half;
      float v = acc[r] + bv;
      if (relu_out) v = fmaxf(v, 0.0f);
      C[(long)m * ldc + n] = v;
    }
  }
}

// ---------------------------------------------------------------------------
// Swizzle f32 activations -> fragment-ready bf16 A layout (fused ReLU, zero pad)
// A 16x32 16-bit layout: lane -> M=lane&15; VGPR0-3: K=8*half+{0..7},
// VGPR4-7: K=16+8*half+{0..7}  (half = lane>>4)
// ---------------------------------------------------------------------------
__global__ void aswz_kernel(const float* __restrict__ A, int lda,
                            unsigned short* __restrict__ Asw,
                            int M, int K, int KC, int relu_in, long total)
{
  long i = (long)blockIdx.x * blockDim.x + threadIdx.x;
  if (i >= total) return;
  const int  e    = (int)(i & 15);
  const int  lane = (int)((i >> 4) & 31);
  const long t    = i >> 9;
  const int  kc   = (int)(t % KC);
  const int  mt   = (int)(t / KC);
  const int  half = lane >> 4, q = lane & 15;
  const int  k = kc * 32 + ((e < 8) ? (8 * half + e) : (16 + 8 * half + (e - 8)));
  const int  m = mt * 16 + q;
  float v = (m < M && k < K) ? A[(long)m * lda + k] : 0.0f;
  if (relu_in) v = fmaxf(v, 0.0f);
  Asw[i] = f2bf(v);
}

// ---------------------------------------------------------------------------
// Swizzle f32 weights -> fragment-ready bf16 B layout (zero pad)
// B 32x16 16-bit layout: lane -> N=lane&15; VGPR0-3: K=8*half+{0..7}, VGPR4-7: K=16+8*half+{0..7}
// ---------------------------------------------------------------------------
__global__ void bswz_kernel(const float* __restrict__ W,
                            unsigned short* __restrict__ Bsw,
                            int K, int Nn, int KC, long total)
{
  long i = (long)blockIdx.x * blockDim.x + threadIdx.x;
  if (i >= total) return;
  const int  e    = (int)(i & 15);
  const int  lane = (int)((i >> 4) & 31);
  const long t    = i >> 9;
  const int  kc   = (int)(t % KC);
  const int  nt   = (int)(t / KC);
  const int  half = lane >> 4, q = lane & 15;
  const int  k = kc * 32 + ((e < 8) ? (8 * half + e) : (16 + 8 * half + (e - 8)));
  const int  n = nt * 16 + q;
  Bsw[i] = (k < K && n < Nn) ? f2bf(W[(long)k * Nn + n]) : (unsigned short)0;
}

// ---------------------------------------------------------------------------
// GCN normalization
// ---------------------------------------------------------------------------
__global__ void deg_init_kernel(float* deg, int n) {
  int i = blockIdx.x * blockDim.x + threadIdx.x;
  if (i < n) deg[i] = 1.0f;                      // self loop
}
__global__ void deg_count_kernel(const int* __restrict__ col, float* deg, int E) {
  int e = blockIdx.x * blockDim.x + threadIdx.x;
  if (e < E) atomicAdd(&deg[col[e]], 1.0f);
}
__global__ void dinv_kernel(const float* __restrict__ deg, float* __restrict__ dinv, int n) {
  int i = blockIdx.x * blockDim.x + threadIdx.x;
  if (i < n) dinv[i] = rsqrtf(deg[i]);           // deg >= 1 always
}

// ---------------------------------------------------------------------------
// Aggregation: out[v] = bias + dinv[v]^2 * T[v], then atomic edge accumulation
// ---------------------------------------------------------------------------
__global__ void agg_init_kernel(const float* __restrict__ T, const float* __restrict__ dinv,
                                const float* __restrict__ bias, float* __restrict__ out,
                                int N_, int F)
{
  long i = (long)blockIdx.x * blockDim.x + threadIdx.x;
  if (i >= (long)N_ * F) return;
  int v = (int)(i / F);
  int f = (int)(i - (long)v * F);
  float d = dinv[v];
  out[i] = bias[f] + d * d * T[i];
}

__global__ __launch_bounds__(256)
void agg_edge_kernel(const int* __restrict__ row, const int* __restrict__ col,
                     const float* __restrict__ dinv, const float* __restrict__ T,
                     float* __restrict__ out, int E, int F)
{
  int e = blockIdx.x * 8 + threadIdx.y;
  if (e >= E) return;
  int u = row[e], v = col[e];
  float c = dinv[u] * dinv[v];
  const float* Tu = T + (long)u * F;
  float* Ov = out + (long)v * F;
  for (int f = threadIdx.x; f < F; f += 32)
    atomicAdd(&Ov[f], c * Tu[f]);                // feature arrays stay L2-resident
}

// ---------------------------------------------------------------------------
// Global max pool of relu(h) per graph (post-relu >= 0 -> int-bit atomicMax valid)
// ---------------------------------------------------------------------------
__global__ void pool_init_kernel(float* g, long total) {
  long i = (long)blockIdx.x * blockDim.x + threadIdx.x;
  if (i < total) g[i] = 0.0f;
}
__global__ void pool_max_kernel(const float* __restrict__ h, const int* __restrict__ batch,
                                float* __restrict__ g, int N_, int F)
{
  long i = (long)blockIdx.x * blockDim.x + threadIdx.x;
  if (i >= (long)N_ * F) return;
  int v = (int)(i / F);
  int f = (int)(i - (long)v * F);
  float val = fmaxf(h[i], 0.0f);                 // relu fused
  atomicMax((int*)&g[(long)batch[v] * F + f], __float_as_int(val));
}

// ---------------------------------------------------------------------------
static inline int cdiv(long a, long b) { return (int)((a + b - 1) / b); }

extern "C" void kernel_launch(void* const* d_in, const int* in_sizes, int n_in,
                              void* d_out, int out_size, void* d_ws, size_t ws_size,
                              hipStream_t stream)
{
  const float* x    = (const float*)d_in[0];
  const int*   ei   = (const int*)  d_in[1];
  const int*   batch= (const int*)  d_in[2];
  const float* W1   = (const float*)d_in[3];  const float* b1  = (const float*)d_in[4];
  const float* W2   = (const float*)d_in[5];  const float* b2  = (const float*)d_in[6];
  const float* W3   = (const float*)d_in[7];  const float* b3  = (const float*)d_in[8];
  const float* Wg1  = (const float*)d_in[9];  const float* bg1 = (const float*)d_in[10];
  const float* Wg2  = (const float*)d_in[11]; const float* bg2 = (const float*)d_in[12];
  float* out = (float*)d_out;
  (void)n_in; (void)out_size; (void)ws_size;

  const int F_IN = 75;
  const int N = in_sizes[0] / F_IN;   // 100000 (multiple of 16)
  const int E = in_sizes[1] / 2;      // 800000
  const int G = 512;
  const int* row = ei;
  const int* col = ei + E;

  // ---- workspace carving ----
  char* ws = (char*)d_ws;
  auto carve = [&](size_t bytes) -> char* {
    char* p = ws; ws += (bytes + 255) & ~(size_t)255; return p;
  };
  float* bufT = (float*)carve((size_t)N * 300 * sizeof(float));
  float* bufA = (float*)carve((size_t)N * 300 * sizeof(float));
  float* deg  = (float*)carve((size_t)N * sizeof(float));
  float* dinv = (float*)carve((size_t)N * sizeof(float));
  float* g    = (float*)carve((size_t)G * 300 * sizeof(float));
  float* gh   = (float*)carve((size_t)G * 1024 * sizeof(float));
  // A swizzle buffer: max Mtiles*KC*512 ushorts = (100000/16)*5*512 (layer 3)
  unsigned short* Asw  = (unsigned short*)carve((size_t)(N / 16) * 5 * 512 * 2);
  unsigned short* W1s  = (unsigned short*)carve((size_t)5  * 3  * 512 * 2);
  unsigned short* W2s  = (unsigned short*)carve((size_t)10 * 3  * 512 * 2);
  unsigned short* W3s  = (unsigned short*)carve((size_t)19 * 5  * 512 * 2);
  unsigned short* Wg1s = (unsigned short*)carve((size_t)64 * 10 * 512 * 2);
  unsigned short* Wg2s = (unsigned short*)carve((size_t)8  * 32 * 512 * 2);

  // ---- weight swizzle to fragment-ready bf16 ----
  auto bswz = [&](const float* W, unsigned short* Bsw, int K, int Nn) {
    const int KC = (K + 31) / 32, Ntiles = (Nn + 15) / 16;
    long total = (long)Ntiles * KC * 512;
    bswz_kernel<<<cdiv(total, 256), 256, 0, stream>>>(W, Bsw, K, Nn, KC, total);
  };
  bswz(W1,  W1s,  75,   75);
  bswz(W2,  W2s,  75,   150);
  bswz(W3,  W3s,  150,  300);
  bswz(Wg1, Wg1s, 300,  1024);
  bswz(Wg2, Wg2s, 1024, 128);

  // ---- GCN normalization ----
  deg_init_kernel<<<cdiv(N, 256), 256, 0, stream>>>(deg, N);
  deg_count_kernel<<<cdiv(E, 256), 256, 0, stream>>>(col, deg, E);
  dinv_kernel<<<cdiv(N, 256), 256, 0, stream>>>(deg, dinv, N);

  auto aswz = [&](const float* A, int lda, int M, int K, int relu_in) {
    const int KC = (K + 31) / 32, Mtiles = (M + 15) / 16;
    long total = (long)Mtiles * KC * 512;
    aswz_kernel<<<cdiv(total, 256), 256, 0, stream>>>(A, lda, Asw, M, K, KC, relu_in, total);
  };
  auto gemm = [&](const unsigned short* Bsw, const float* bias, float* C, int ldc,
                  int M, int Nn, int K, int relu_out) {
    const int KC = (K + 31) / 32, Mtiles = (M + 15) / 16, Ntiles = (Nn + 15) / 16;
    dim3 grid(cdiv(Mtiles, 4), Ntiles);
    wmma_gemm_kernel<<<grid, dim3(32, 4), 0, stream>>>(Asw, Bsw, bias, C, ldc,
                                                       Mtiles, Ntiles, KC, Nn, relu_out);
  };
  auto agg = [&](const float* T, const float* bias, float* O, int F) {
    long total = (long)N * F;
    agg_init_kernel<<<cdiv(total, 256), 256, 0, stream>>>(T, dinv, bias, O, N, F);
    agg_edge_kernel<<<cdiv(E, 8), dim3(32, 8), 0, stream>>>(row, col, dinv, T, O, E, F);
  };

  // ---- layer 1: T = x @ W1 ; H1 = agg(T) + b1 (ReLU deferred to next A-load) ----
  aswz(x, 75, N, 75, 0);
  gemm(W1s, nullptr, bufT, 75, N, 75, 75, 0);
  agg(bufT, b1, bufA, 75);
  // ---- layer 2 ----
  aswz(bufA, 75, N, 75, 1);
  gemm(W2s, nullptr, bufT, 150, N, 150, 75, 0);
  agg(bufT, b2, bufA, 150);
  // ---- layer 3 ----
  aswz(bufA, 150, N, 150, 1);
  gemm(W3s, nullptr, bufT, 300, N, 300, 150, 0);
  agg(bufT, b3, bufA, 300);

  // ---- global max pool (relu fused) ----
  {
    long gt = (long)G * 300;
    pool_init_kernel<<<cdiv(gt, 256), 256, 0, stream>>>(g, gt);
    long ht = (long)N * 300;
    pool_max_kernel<<<cdiv(ht, 256), 256, 0, stream>>>(bufA, batch, g, N, 300);
  }

  // ---- dense head ----
  aswz(g, 300, G, 300, 0);
  gemm(Wg1s, bg1, gh, 1024, G, 1024, 300, 1);
  aswz(gh, 1024, G, 1024, 0);
  gemm(Wg2s, bg2, out, 128, G, 128, 1024, 0);
}
